// UncertaintyEstimatorIterativeNarrowMambaBlock_14164802142699
// MI455X (gfx1250) — matrix operand order repaired and checked by reading, not compiled, not run on hardware
//
#include <hip/hip_runtime.h>
#include <hip/hip_bf16.h>
#include <math.h>

// ---------------------------------------------------------------------------
// UncertaintyEstimatorIterativeNarrowMambaBlock for MI455X (gfx1250, wave32)
//
// Sizes: B=2, L=3 -> bl=6 images, C=64, H=W=32 (P=1024 pixels), DI=128,
// N=16 states, DT_RANK=4, K=4 scan directions. M (GEMM rows) = 6*1024 = 6144.
//
// Strategy:
//  * All dense / conv matmuls run on v_wmma_f32_16x16x32_f16, one wave per
//    16x16 output tile.  Activations live in f16 row-major [row][feat];
//    weights are pre-packed into the exact B-fragment lane layout so a wave
//    feeds WMMA with two b128 loads (A) + two b128 loads (B) per k-chunk.
//  * k_gemm is fully template-specialized per call site (lda/ldo, tile counts,
//    k-chunk count, conv Cin, scan direction, epilogue flags) so the k-loop
//    unrolls into straight-line load+wmma sequences with no runtime div/mod.
//  * Conv inputs carry a zeroed 1-pixel halo (34x34 per image): the im2col
//    tap shift becomes a compile-time immediate offset on the fragment load —
//    no boundary branches, no zero-fill path, no EXEC manipulation in-loop.
//  * The selective scan keeps 16 states in registers, one (k,img,d)
//    recurrence per lane (3072 lanes), streaming dt/B/C from L2.
// ---------------------------------------------------------------------------

typedef __attribute__((ext_vector_type(16))) _Float16 v16h;
typedef __attribute__((ext_vector_type(8)))  _Float16 v8h;
typedef __attribute__((ext_vector_type(8)))  float    v8f;

#define NIMG   6
#define PIX    1024
#define MROWS  (NIMG * PIX)   // 6144
#define MTILES 384            // 6144 / 16
#define HW     34             // 32 + 1-pixel halo on each side
#define HPIX   (HW * HW)      // 1156

// direction permutations ------------------------------------------------------
// xs_k[d][l] = xc[d][perm_pix(k,l)]
__device__ __forceinline__ int perm_pix(int k, int l) {
  int j = (k & 2) ? (1023 - l) : l;
  return (k & 1) ? (((j & 31) << 5) | (j >> 5)) : j;
}
// sequence position of direction k that lands on pixel p
__device__ __forceinline__ int inv_perm(int k, int p) {
  int j = (k & 1) ? (((p & 31) << 5) | (p >> 5)) : p;
  return (k & 2) ? (1023 - j) : j;
}

// ---------------------------------------------------------------------------
// Weight packing into WMMA B-fragment layout.
// B 32x16 f16 packing (this code's contiguous k order, same as A): lane -> n = lane&15, kb = (lane<16)?0:16,
// element e (0..15) -> K = kc*32 + kb + e.
// Fragment buffer: [ntile][kchunk][lane][16] halves.
// ---------------------------------------------------------------------------
__global__ void k_pack_dense(const float* __restrict__ w, _Float16* __restrict__ bp,
                             int Ntiles, int kchunks, int Nreal, int Kreal) {
  int idx = blockIdx.x * blockDim.x + threadIdx.x;
  int total = Ntiles * kchunks * 512;
  if (idx >= total) return;
  int tile = idx >> 9, r = idx & 511;
  int lane = r >> 4, e = r & 15;
  int nt = tile / kchunks, kc = tile % kchunks;
  int n  = nt * 16 + (lane & 15);
  int kb = (lane & 16) ? 16 : 0;
  int K  = kc * 32 + kb + e;
  float v = (n < Nreal && K < Kreal) ? w[(size_t)n * Kreal + K] : 0.f;
  bp[idx] = (_Float16)v;
}

// conv weights [cout][cin][3][3] with im2col K = tap*cin + ci
__global__ void k_pack_conv(const float* __restrict__ w, _Float16* __restrict__ bp,
                            int Ntiles, int cin) {
  int kchunks = (cin * 9) >> 5;
  int idx = blockIdx.x * blockDim.x + threadIdx.x;
  int total = Ntiles * kchunks * 512;
  if (idx >= total) return;
  int tile = idx >> 9, r = idx & 511;
  int lane = r >> 4, e = r & 15;
  int nt = tile / kchunks, kc = tile % kchunks;
  int n  = nt * 16 + (lane & 15);
  int kb = (lane & 16) ? 16 : 0;
  int K  = kc * 32 + kb + e;
  int tap = K / cin, ci = K - tap * cin;
  bp[idx] = (_Float16)w[((size_t)n * cin + ci) * 9 + tap];
}

// zero-fill an f16 buffer (for conv halos)
__global__ void k_zero_h(_Float16* __restrict__ p, int count) {
  int idx = blockIdx.x * blockDim.x + threadIdx.x;
  if (idx < count) p[idx] = (_Float16)0.f;
}

// ---------------------------------------------------------------------------
// Fully specialized WMMA GEMM: out[6144][*] = A(f16) * Wpacked.
// AMODE: 0 plain rows [row][LDA], 1 conv over halo image [img][34*34][LDA]
//        (AUX = Cin), 2 direction-permuted rows (AUX = direction k).
// ACT: 0 none, 1 exact GELU (applied before addsrc).
// HB/HA/OF/OH: bias / addsrc / f32 out / f16 out enables.
// ---------------------------------------------------------------------------
template <int AMODE, int LDA, int LDO, int NTILES, int KCHUNKS, int AUX, int ACT,
          bool HB, bool HA, bool OF, bool OH>
__global__ void k_gemm(const _Float16* __restrict__ A,
                       const _Float16* __restrict__ Bp,
                       const float* __restrict__ bias,
                       const float* __restrict__ addsrc,
                       float* __restrict__ outF, _Float16* __restrict__ outH) {
  int wid  = (blockIdx.x * blockDim.x + threadIdx.x) >> 5;
  int lane = threadIdx.x & 31;
  if (wid >= MTILES * NTILES) return;
  int mt = wid / NTILES, nt = wid % NTILES;   // compile-time divisor
  int row = mt * 16 + (lane & 15);
  int kb  = (lane & 16) ? 16 : 0;  // this code's A k order: lanes 16-31 hold K=16..31

  const _Float16* abase;
  if constexpr (AMODE == 0) {
    abase = A + (size_t)row * LDA;
  } else if constexpr (AMODE == 2) {
    int img = row >> 10;
    int l = row & 1023;
    abase = A + (size_t)((img << 10) + perm_pix(AUX, l)) * LDA;
  } else {  // conv over halo image: center pixel of this row
    int img = row >> 10;
    int p = row & 1023;
    abase = A + (size_t)(img * HPIX + ((p >> 5) + 1) * HW + (p & 31) + 1) * LDA;
  }
  const _Float16* bp = Bp + (size_t)(nt * KCHUNKS) * 512 + lane * 16;

  v8f acc = {};
#pragma unroll
  for (int kc = 0; kc < KCHUNKS; ++kc) {
    const _Float16* ap;
    if constexpr (AMODE == 1) {
      constexpr int CIN = AUX;                 // 64 or 192, multiple of 32
      // chunk lies fully inside one tap; tap shift is a compile-time offset
      int kk  = kc * 32;
      int tap = kk / CIN;
      int c0  = kk - tap * CIN;
      int dy  = tap / 3 - 1, dx = tap % 3 - 1;
      ap = abase + (dy * HW + dx) * LDA + c0;  // constant fold -> load imm offset
    } else {
      ap = abase + kc * 32;
    }
    v8h lo = *(const v8h*)(ap + kb);        // K = kb .. kb+7
    v8h hi = *(const v8h*)(ap + kb + 8);    // K = kb+8 .. kb+15
    v16h a;
#pragma unroll
    for (int i = 0; i < 8; ++i) { a[i] = lo[i]; a[8 + i] = hi[i]; }
    v16h bfr = *(const v16h*)(bp + (size_t)kc * 512);
    acc = __builtin_amdgcn_wmma_f32_16x16x32_f16(false, a, false, bfr,
                                                 (short)0, acc, false, false);
  }

  int col   = nt * 16 + (lane & 15);
  int rbase = mt * 16 + ((lane & 16) ? 8 : 0);
  float bv = 0.f;
  if constexpr (HB) bv = bias[col];
#pragma unroll
  for (int r = 0; r < 8; ++r) {
    size_t o = (size_t)(rbase + r) * LDO + col;
    float v = acc[r] + bv;
    if constexpr (ACT == 1) v = 0.5f * v * (1.f + erff(v * 0.70710678f));  // exact GELU
    if constexpr (HA) v += addsrc[o];
    if constexpr (OF) outF[o] = v;
    if constexpr (OH) outH[o] = (_Float16)v;
  }
}

// ---------------------------------------------------------------------------
// Elementwise / reduction glue
// ---------------------------------------------------------------------------

// per-pixel channel LayerNorm on the frames + concat [dz | ln(frame) | sigma]
__global__ void k_build_x(const float* __restrict__ seq, const float* __restrict__ dz,
                          const float* __restrict__ sg, const float* __restrict__ ng,
                          const float* __restrict__ nb, _Float16* __restrict__ xcat) {
  int idx = blockIdx.x * blockDim.x + threadIdx.x;
  if (idx >= MROWS) return;
  int img = idx >> 10, p = idx & 1023;
  int b = img / 3;
  const float* fr  = seq + (size_t)img * 64 * PIX + p;
  const float* dzp = dz  + (size_t)b   * 64 * PIX + p;
  const float* sgp = sg  + (size_t)b   * 64 * PIX + p;
  float m = 0.f;
  for (int c = 0; c < 64; ++c) m += fr[c * PIX];
  m *= (1.f / 64.f);
  float v = 0.f;
  for (int c = 0; c < 64; ++c) { float t = fr[c * PIX] - m; v += t * t; }
  float rs = rsqrtf(v * (1.f / 64.f) + 1e-6f);
  _Float16* rowp = xcat + (size_t)idx * 192;
  for (int c = 0; c < 64; ++c) {
    rowp[c]       = (_Float16)dzp[c * PIX];
    rowp[64 + c]  = (_Float16)(((fr[c * PIX] - m) * rs) * ng[c] + nb[c]);
    rowp[128 + c] = (_Float16)sgp[c * PIX];
  }
}

// GroupNorm(1) statistics over one image, f16 input, [img][2] = mean, rstd
__global__ void k_gn_stats(const _Float16* __restrict__ x, float* __restrict__ stats,
                           int nch) {
  __shared__ float s1[256], s2[256];
  int img = blockIdx.x, t = threadIdx.x;
  size_t n = (size_t)nch * PIX;
  const _Float16* p = x + (size_t)img * n;
  float a = 0.f, b = 0.f;
  for (size_t i = t; i < n; i += 256) { float v = (float)p[i]; a += v; b += v * v; }
  s1[t] = a; s2[t] = b; __syncthreads();
  for (int s = 128; s > 0; s >>= 1) {
    if (t < s) { s1[t] += s1[t + s]; s2[t] += s2[t + s]; }
    __syncthreads();
  }
  if (t == 0) {
    float inv = 1.f / (float)n;
    float m = s1[0] * inv, var = s2[0] * inv - m * m;
    stats[img * 2] = m;
    stats[img * 2 + 1] = rsqrtf(var + 1e-5f);
  }
}

// apply GroupNorm affine + SiLU, dense [img*1024][NCH] -> halo [img][34*34][NCH]
template <int NCH>
__global__ void k_gn_silu_halo(const _Float16* __restrict__ x, _Float16* __restrict__ y,
                               const float* __restrict__ g, const float* __restrict__ b,
                               const float* __restrict__ stats) {
  int idx = blockIdx.x * blockDim.x + threadIdx.x;
  if (idx >= MROWS * NCH) return;
  int img = idx / (PIX * NCH);
  int rem = idx - img * (PIX * NCH);
  int p = rem / NCH, c = rem - (rem / NCH) * NCH;
  float m = stats[img * 2], rs = stats[img * 2 + 1];
  float v = ((float)x[idx] - m) * rs * g[c] + b[c];
  v = v / (1.f + __expf(-v));
  int hp = ((p >> 5) + 1) * HW + (p & 31) + 1;
  y[((size_t)img * HPIX + hp) * NCH + c] = (_Float16)v;
}

// row-wise LayerNorm over nf features -> f16
__global__ void k_ln_rows(const float* __restrict__ x, const float* __restrict__ g,
                          const float* __restrict__ b, _Float16* __restrict__ out, int nf) {
  int row = blockIdx.x * blockDim.x + threadIdx.x;
  if (row >= MROWS) return;
  const float* p = x + (size_t)row * nf;
  float m = 0.f;
  for (int i = 0; i < nf; ++i) m += p[i];
  m /= (float)nf;
  float v = 0.f;
  for (int i = 0; i < nf; ++i) { float t = p[i] - m; v += t * t; }
  float rs = rsqrtf(v / (float)nf + 1e-5f);
  _Float16* o = out + (size_t)row * nf;
  for (int i = 0; i < nf; ++i) o[i] = (_Float16)((p[i] - m) * rs * g[i] + b[i]);
}

// depthwise 3x3 conv on xc = xz[:, :128] (pixel-major, ld 256) + SiLU -> f16
__global__ void k_dwconv(const float* __restrict__ xz, const float* __restrict__ w,
                         const float* __restrict__ b, _Float16* __restrict__ out) {
  int idx = blockIdx.x * blockDim.x + threadIdx.x;
  if (idx >= MROWS * 128) return;
  int d = idx & 127, row = idx >> 7;
  int img = row >> 10, p = row & 1023;
  int py = p >> 5, px = p & 31;
  float s = b[d];
#pragma unroll
  for (int t = 0; t < 9; ++t) {
    int yy = py + t / 3 - 1, xx = px + t % 3 - 1;
    if ((unsigned)yy < 32u && (unsigned)xx < 32u)
      s += w[d * 9 + t] * xz[(size_t)((img << 10) + (yy << 5) + xx) * 256 + d];
  }
  s = s / (1.f + __expf(-s));
  out[(size_t)row * 128 + d] = (_Float16)s;
}

// dt = softplus(dts_raw @ dt_proj_w^T + dt_proj_b), written [k][img][d][l]
__global__ void k_dtproj(const float* __restrict__ xdbl, const float* __restrict__ dtw,
                         const float* __restrict__ dtb, float* __restrict__ dts) {
  int idx = blockIdx.x * blockDim.x + threadIdx.x;
  if (idx >= 4 * MROWS) return;
  int k = idx / MROWS, row = idx % MROWS;
  int img = row >> 10, l = row & 1023;
  const float* xr = xdbl + ((size_t)k * MROWS + row) * 48;
  float r0 = xr[0], r1 = xr[1], r2 = xr[2], r3 = xr[3];
  for (int d = 0; d < 128; ++d) {
    const float* w = dtw + ((size_t)k * 128 + d) * 4;
    float s = r0 * w[0] + r1 * w[1] + r2 * w[2] + r3 * w[3] + dtb[k * 128 + d];
    s = (s > 20.f) ? s : log1pf(__expf(s));
    dts[(((size_t)k * NIMG + img) * 128 + d) * PIX + l] = s;
  }
}

// selective scan: one (k,img,d) recurrence per lane, 16 states in registers
__global__ void k_scan(const float* __restrict__ dts, const float* __restrict__ xdbl,
                       const _Float16* __restrict__ xch, const float* __restrict__ A_logs,
                       const float* __restrict__ Ds, float* __restrict__ ys) {
  int t = blockIdx.x * blockDim.x + threadIdx.x;
  if (t >= 4 * NIMG * 128) return;
  int k = t / (NIMG * 128);
  int rem = t % (NIMG * 128);
  int img = rem / 128, d = rem % 128;
  float An[16], h[16];
#pragma unroll
  for (int n = 0; n < 16; ++n) {
    An[n] = -__expf(A_logs[((size_t)k * 128 + d) * 16 + n]);
    h[n] = 0.f;
  }
  float Dv = Ds[k * 128 + d];
  const float* dtp = dts + (((size_t)k * NIMG + img) * 128 + d) * PIX;
  const float* xr0 = xdbl + ((size_t)k * MROWS + (img << 10)) * 48;
  float* yo = ys + (((size_t)k * NIMG + img) * 128 + d) * PIX;
  for (int l = 0; l < PIX; ++l) {
    float dt = dtp[l];
    float u  = (float)xch[(size_t)((img << 10) + perm_pix(k, l)) * 128 + d];
    const float* xr = xr0 + (size_t)l * 48;
    float dtu = dt * u, y = 0.f;
#pragma unroll
    for (int n = 0; n < 16; ++n) {
      h[n] = __expf(dt * An[n]) * h[n] + dtu * xr[4 + n];   // B at cols 4..19
      y += h[n] * xr[20 + n];                               // C at cols 20..35
    }
    yo[l] = y + u * Dv;
  }
}

// combine the four direction outputs back to pixel space
__global__ void k_combine(const float* __restrict__ ys, float* __restrict__ ysum) {
  int idx = blockIdx.x * blockDim.x + threadIdx.x;
  if (idx >= MROWS * 128) return;
  int d = idx & 127, row = idx >> 7;
  int img = row >> 10, p = row & 1023;
  float s = 0.f;
#pragma unroll
  for (int k = 0; k < 4; ++k) {
    int l = inv_perm(k, p);
    s += ys[(((size_t)k * NIMG + img) * 128 + d) * PIX + l];
  }
  ysum[(size_t)row * 128 + d] = s;
}

// out_norm LayerNorm + SiLU(z) gate -> f16 for out_proj
__global__ void k_gate(const float* __restrict__ ysum, const float* __restrict__ xz,
                       const float* __restrict__ g, const float* __restrict__ b,
                       _Float16* __restrict__ yh) {
  int row = blockIdx.x * blockDim.x + threadIdx.x;
  if (row >= MROWS) return;
  const float* p = ysum + (size_t)row * 128;
  float m = 0.f;
  for (int i = 0; i < 128; ++i) m += p[i];
  m *= (1.f / 128.f);
  float v = 0.f;
  for (int i = 0; i < 128; ++i) { float t = p[i] - m; v += t * t; }
  float rs = rsqrtf(v * (1.f / 128.f) + 1e-5f);
  for (int d = 0; d < 128; ++d) {
    float val = (p[d] - m) * rs * g[d] + b[d];
    float z = xz[(size_t)row * 256 + 128 + d];
    val *= z / (1.f + __expf(-z));
    yh[(size_t)row * 128 + d] = (_Float16)val;
  }
}

// [row][c] pixel-major -> [img][c][p] NCHW output
__global__ void k_final(const float* __restrict__ x3, float* __restrict__ out) {
  int idx = blockIdx.x * blockDim.x + threadIdx.x;
  if (idx >= MROWS * 64) return;
  int c = idx & 63, row = idx >> 6;
  int img = row >> 10, p = row & 1023;
  out[((size_t)img * 64 + c) * PIX + p] = x3[(size_t)row * 64 + c];
}

// ---------------------------------------------------------------------------
// Host launcher
// ---------------------------------------------------------------------------
extern "C" void kernel_launch(void* const* d_in, const int* in_sizes, int n_in,
                              void* d_out, int out_size, void* d_ws, size_t ws_size,
                              hipStream_t stream) {
  (void)in_sizes; (void)n_in; (void)out_size; (void)ws_size;

  const float* seq       = (const float*)d_in[0];
  const float* dz        = (const float*)d_in[1];
  const float* sg        = (const float*)d_in[2];
  const float* norm_g    = (const float*)d_in[3];
  const float* norm_b    = (const float*)d_in[4];
  const float* gn1_g     = (const float*)d_in[5];
  const float* gn1_b     = (const float*)d_in[6];
  const float* conv1_w   = (const float*)d_in[7];
  const float* conv1_b   = (const float*)d_in[8];
  const float* gn2_g     = (const float*)d_in[9];
  const float* gn2_b     = (const float*)d_in[10];
  const float* conv2_w   = (const float*)d_in[11];
  const float* conv2_b   = (const float*)d_in[12];
  const float* skip_w    = (const float*)d_in[13];
  const float* skip_b    = (const float*)d_in[14];
  const float* ln1_g     = (const float*)d_in[15];
  const float* ln1_b     = (const float*)d_in[16];
  const float* ln2_g     = (const float*)d_in[17];
  const float* ln2_b     = (const float*)d_in[18];
  const float* in_proj_w = (const float*)d_in[19];
  const float* dwconv_w  = (const float*)d_in[20];
  const float* dwconv_b  = (const float*)d_in[21];
  const float* x_proj_w  = (const float*)d_in[22];
  const float* dt_proj_w = (const float*)d_in[23];
  const float* dt_proj_b = (const float*)d_in[24];
  const float* A_logs    = (const float*)d_in[25];
  const float* Ds        = (const float*)d_in[26];
  const float* onorm_g   = (const float*)d_in[27];
  const float* onorm_b   = (const float*)d_in[28];
  const float* out_proj_w= (const float*)d_in[29];
  const float* fc1_w     = (const float*)d_in[30];
  const float* fc1_b     = (const float*)d_in[31];
  const float* fc2_w     = (const float*)d_in[32];
  const float* fc2_b     = (const float*)d_in[33];

  // workspace bump allocator (256B aligned)
  char* wsb = (char*)d_ws;
  size_t off = 0;
  auto alloc = [&](size_t bytes) -> void* {
    void* p = wsb + off;
    off = (off + bytes + 255) & ~(size_t)255;
    return p;
  };

  _Float16* xcat_h  = (_Float16*)alloc((size_t)MROWS * 192 * 2);
  _Float16* act1_h  = (_Float16*)alloc((size_t)NIMG * HPIX * 192 * 2); // halo
  _Float16* hdn1_h  = (_Float16*)alloc((size_t)MROWS * 64  * 2);
  _Float16* act2_h  = (_Float16*)alloc((size_t)NIMG * HPIX * 64 * 2);  // halo
  float*    xres    = (float*)   alloc((size_t)MROWS * 64  * 4);
  _Float16* xn_h    = (_Float16*)alloc((size_t)MROWS * 64  * 2);
  float*    xz      = (float*)   alloc((size_t)MROWS * 256 * 4);
  _Float16* xch_h   = (_Float16*)alloc((size_t)MROWS * 128 * 2);
  float*    xdbl    = (float*)   alloc((size_t)4 * MROWS * 48 * 4);
  float*    dts     = (float*)   alloc((size_t)4 * MROWS * 128 * 4);
  float*    ys      = (float*)   alloc((size_t)4 * MROWS * 128 * 4);
  float*    ysum    = (float*)   alloc((size_t)MROWS * 128 * 4);
  _Float16* y_h     = (_Float16*)alloc((size_t)MROWS * 128 * 2);
  float*    x2      = (float*)   alloc((size_t)MROWS * 64  * 4);
  _Float16* xn2_h   = (_Float16*)alloc((size_t)MROWS * 64  * 2);
  _Float16* mlp_h   = (_Float16*)alloc((size_t)MROWS * 64  * 2);
  float*    x3      = (float*)   alloc((size_t)MROWS * 64  * 4);
  float*    stats1  = (float*)   alloc(NIMG * 2 * 4);
  float*    stats2  = (float*)   alloc(NIMG * 2 * 4);
  _Float16* bp_in   = (_Float16*)alloc((size_t)16 * 2  * 512 * 2); // in_proj  N=256 K=64
  _Float16* bp_skip = (_Float16*)alloc((size_t)4  * 6  * 512 * 2); // skip     N=64  K=192
  _Float16* bp_c1   = (_Float16*)alloc((size_t)4  * 54 * 512 * 2); // conv1    N=64  K=1728
  _Float16* bp_c2   = (_Float16*)alloc((size_t)4  * 18 * 512 * 2); // conv2    N=64  K=576
  _Float16* bp_xp   = (_Float16*)alloc((size_t)4 * 3 * 4 * 512 * 2); // x_proj x4, N=48(pad36) K=128
  _Float16* bp_op   = (_Float16*)alloc((size_t)4  * 4  * 512 * 2); // out_proj N=64  K=128
  _Float16* bp_f1   = (_Float16*)alloc((size_t)4  * 2  * 512 * 2); // fc1      N=64  K=64
  _Float16* bp_f2   = (_Float16*)alloc((size_t)4  * 2  * 512 * 2); // fc2      N=64  K=64

  auto grid1 = [](int n, int b) { return dim3((unsigned)((n + b - 1) / b)); };
  auto ggrid = [](int Ntiles) { return dim3((unsigned)((MTILES * Ntiles + 7) / 8)); };

  // ---- weight packing + halo zeroing ---------------------------------------
  k_pack_dense<<<grid1(16 * 2 * 512, 256), 256, 0, stream>>>(in_proj_w, bp_in, 16, 2, 256, 64);
  k_pack_dense<<<grid1(4 * 6 * 512, 256), 256, 0, stream>>>(skip_w, bp_skip, 4, 6, 64, 192);
  k_pack_conv <<<grid1(4 * 54 * 512, 256), 256, 0, stream>>>(conv1_w, bp_c1, 4, 192);
  k_pack_conv <<<grid1(4 * 18 * 512, 256), 256, 0, stream>>>(conv2_w, bp_c2, 4, 64);
  for (int k = 0; k < 4; ++k)
    k_pack_dense<<<grid1(3 * 4 * 512, 256), 256, 0, stream>>>(
        x_proj_w + (size_t)k * 36 * 128, bp_xp + (size_t)k * 3 * 4 * 512, 3, 4, 36, 128);
  k_pack_dense<<<grid1(4 * 4 * 512, 256), 256, 0, stream>>>(out_proj_w, bp_op, 4, 4, 64, 128);
  k_pack_dense<<<grid1(4 * 2 * 512, 256), 256, 0, stream>>>(fc1_w, bp_f1, 4, 2, 64, 64);
  k_pack_dense<<<grid1(4 * 2 * 512, 256), 256, 0, stream>>>(fc2_w, bp_f2, 4, 2, 64, 64);
  k_zero_h<<<grid1(NIMG * HPIX * 192, 256), 256, 0, stream>>>(act1_h, NIMG * HPIX * 192);
  k_zero_h<<<grid1(NIMG * HPIX * 64, 256), 256, 0, stream>>>(act2_h, NIMG * HPIX * 64);

  // ---- stem: LN2d + concat, GN+SiLU, conv res block -------------------------
  k_build_x<<<grid1(MROWS, 256), 256, 0, stream>>>(seq, dz, sg, norm_g, norm_b, xcat_h);
  k_gn_stats<<<NIMG, 256, 0, stream>>>(xcat_h, stats1, 192);
  k_gn_silu_halo<192><<<grid1(MROWS * 192, 256), 256, 0, stream>>>(
      xcat_h, act1_h, gn1_g, gn1_b, stats1);
  // conv1: implicit GEMM on WMMA  <AMODE,LDA,LDO,NT,KC,AUX,ACT, HB,HA,OF,OH>
  k_gemm<1, 192, 64, 4, 54, 192, 0, true, false, false, true>
      <<<ggrid(4), 256, 0, stream>>>(act1_h, bp_c1, conv1_b, nullptr, nullptr, hdn1_h);
  k_gn_stats<<<NIMG, 256, 0, stream>>>(hdn1_h, stats2, 64);
  k_gn_silu_halo<64><<<grid1(MROWS * 64, 256), 256, 0, stream>>>(
      hdn1_h, act2_h, gn2_g, gn2_b, stats2);
  // skip 1x1 then conv2 accumulating into xres
  k_gemm<0, 192, 64, 4, 6, 0, 0, true, false, true, false>
      <<<ggrid(4), 256, 0, stream>>>(xcat_h, bp_skip, skip_b, nullptr, xres, nullptr);
  k_gemm<1, 64, 64, 4, 18, 64, 0, true, true, true, false>
      <<<ggrid(4), 256, 0, stream>>>(act2_h, bp_c2, conv2_b, xres, xres, nullptr);

  // ---- SS2D -----------------------------------------------------------------
  k_ln_rows<<<grid1(MROWS, 256), 256, 0, stream>>>(xres, ln1_g, ln1_b, xn_h, 64);
  k_gemm<0, 64, 256, 16, 2, 0, 0, false, false, true, false>
      <<<ggrid(16), 256, 0, stream>>>(xn_h, bp_in, nullptr, nullptr, xz, nullptr);
  k_dwconv<<<grid1(MROWS * 128, 256), 256, 0, stream>>>(xz, dwconv_w, dwconv_b, xch_h);
  // 4 direction-permuted x_proj GEMMs
  k_gemm<2, 128, 48, 3, 4, 0, 0, false, false, true, false><<<ggrid(3), 256, 0, stream>>>(
      xch_h, bp_xp + (size_t)0 * 3 * 4 * 512, nullptr, nullptr, xdbl + (size_t)0 * MROWS * 48, nullptr);
  k_gemm<2, 128, 48, 3, 4, 1, 0, false, false, true, false><<<ggrid(3), 256, 0, stream>>>(
      xch_h, bp_xp + (size_t)1 * 3 * 4 * 512, nullptr, nullptr, xdbl + (size_t)1 * MROWS * 48, nullptr);
  k_gemm<2, 128, 48, 3, 4, 2, 0, false, false, true, false><<<ggrid(3), 256, 0, stream>>>(
      xch_h, bp_xp + (size_t)2 * 3 * 4 * 512, nullptr, nullptr, xdbl + (size_t)2 * MROWS * 48, nullptr);
  k_gemm<2, 128, 48, 3, 4, 3, 0, false, false, true, false><<<ggrid(3), 256, 0, stream>>>(
      xch_h, bp_xp + (size_t)3 * 3 * 4 * 512, nullptr, nullptr, xdbl + (size_t)3 * MROWS * 48, nullptr);
  k_dtproj<<<grid1(4 * MROWS, 256), 256, 0, stream>>>(xdbl, dt_proj_w, dt_proj_b, dts);
  k_scan<<<grid1(4 * NIMG * 128, 256), 256, 0, stream>>>(dts, xdbl, xch_h, A_logs, Ds, ys);
  k_combine<<<grid1(MROWS * 128, 256), 256, 0, stream>>>(ys, ysum);
  k_gate<<<grid1(MROWS, 256), 256, 0, stream>>>(ysum, xz, onorm_g, onorm_b, y_h);
  k_gemm<0, 128, 64, 4, 4, 0, 0, false, true, true, false>
      <<<ggrid(4), 256, 0, stream>>>(y_h, bp_op, nullptr, xres, x2, nullptr);

  // ---- MLP ------------------------------------------------------------------
  k_ln_rows<<<grid1(MROWS, 256), 256, 0, stream>>>(x2, ln2_g, ln2_b, xn2_h, 64);
  k_gemm<0, 64, 64, 4, 2, 0, 1, true, false, false, true>
      <<<ggrid(4), 256, 0, stream>>>(xn2_h, bp_f1, fc1_b, nullptr, nullptr, mlp_h);  // GELU
  k_gemm<0, 64, 64, 4, 2, 0, 0, true, true, true, false>
      <<<ggrid(4), 256, 0, stream>>>(mlp_h, bp_f2, fc2_b, x2, x3, nullptr);

  // ---- output transpose -----------------------------------------------------
  k_final<<<grid1(MROWS * 64, 256), 256, 0, stream>>>(x3, (float*)d_out);
}
